// Self_Attention_Fusion_56916906607025
// MI455X (gfx1250) — matrix operand order repaired and checked by reading, not compiled, not run on hardware
//
#include <hip/hip_runtime.h>
#include <hip/hip_bf16.h>

typedef __attribute__((ext_vector_type(16))) _Float16 v16h;
typedef __attribute__((ext_vector_type(8)))  float    v8f;

#define WMMA_F16(a,b,c) __builtin_amdgcn_wmma_f32_16x16x32_f16(false,(a),false,(b),(short)0,(c),false,false)

#ifndef __has_builtin
#define __has_builtin(x) 0
#endif
#if __has_builtin(__builtin_amdgcn_tensor_load_to_lds)
#define USE_TDM 1
#else
#define USE_TDM 0
#endif

constexpr int Bn = 16;     // batch
constexpr int Cc = 256;    // channels
constexpr int Dq = 64;     // query/key dim
constexpr int Nn = 4096;   // pixels (64*64)
constexpr int PSTRIDE = 72;  // P LDS row stride (halves): 144B, 16B aligned, conflict-free
constexpr int IW = 32;     // i-tile width per workgroup (attend)
constexpr int JC = 64;     // j chunk per iteration (attend)

__device__ __forceinline__ int laneid() { return (int)(threadIdx.x & 31); }

// ---- WMMA operand loaders (wave32; contiguous k = 16*(lane/16) + s in both A and B) ----

// A (16x32 f16): A[row][k] = src[(row0+row)*ld + k0 + k], src f16 row-major.
__device__ __forceinline__ v16h load_a_h(const _Float16* p0, int ld, int row0, int k0) {
  const int lane = laneid();
  const int kb = (lane >> 4) * 16;
  const _Float16* p = p0 + (size_t)(row0 + (lane & 15)) * ld + k0;
  v16h a;
#pragma unroll
  for (int h = 0; h < 8; ++h) a[h] = p[kb + h];
#pragma unroll
  for (int h = 0; h < 8; ++h) a[8 + h] = p[8 + kb + h];
  return a;
}

// B (32x16 f16): B[k][col] = src[(col0+col)*ld + k0 + k]  (K contiguous per column)
__device__ __forceinline__ v16h load_b_h_kmajor(const _Float16* p0, int ld, int col0, int k0) {
  const int lane = laneid();
  const _Float16* p = p0 + (size_t)(col0 + (lane & 15)) * ld + k0 + ((lane >> 4) * 16);
  v16h b;
#pragma unroll
  for (int h = 0; h < 16; ++h) b[h] = p[h];
  return b;
}

// A from f32 X^T: A[row][k] = X[(k0+k)*n + row0+row]  (strided gather + cvt to f16)
__device__ __forceinline__ v16h load_a_xT(const float* X, int n, int row0, int k0) {
  const int lane = laneid();
  const int kb = (lane >> 4) * 16;
  const int j = row0 + (lane & 15);
  v16h a;
#pragma unroll
  for (int h = 0; h < 8; ++h) a[h] = (_Float16)X[(size_t)(k0 + kb + h) * n + j];
#pragma unroll
  for (int h = 0; h < 8; ++h) a[8 + h] = (_Float16)X[(size_t)(k0 + 8 + kb + h) * n + j];
  return a;
}

// B from f32 W^T: B[k][col] = W[(col0+col)*ldw + k0 + k]  (contiguous + cvt)
__device__ __forceinline__ v16h load_b_wT(const float* W, int ldw, int col0, int k0) {
  const int lane = laneid();
  const float* p = W + (size_t)(col0 + (lane & 15)) * ldw + k0 + ((lane >> 4) * 16);
  v16h b;
#pragma unroll
  for (int h = 0; h < 16; ++h) b[h] = (_Float16)p[h];
  return b;
}

// A from f32 row-major W: A[row][k] = W[(row0+row)*ldw + k0 + k]
__device__ __forceinline__ v16h load_a_w(const float* W, int ldw, int row0, int k0) {
  const int lane = laneid();
  const int kb = (lane >> 4) * 16;
  const float* p = W + (size_t)(row0 + (lane & 15)) * ldw + k0;
  v16h a;
#pragma unroll
  for (int h = 0; h < 8; ++h) a[h] = (_Float16)p[kb + h];
#pragma unroll
  for (int h = 0; h < 8; ++h) a[8 + h] = (_Float16)p[8 + kb + h];
  return a;
}

// B from f32 X: B[k][col] = X[(k0+k)*n + col0+col]  (strided gather + cvt)
__device__ __forceinline__ v16h load_b_x(const float* X, int n, int col0, int k0) {
  const int lane = laneid();
  const int kb = (lane >> 4) * 16;
  const int j = col0 + (lane & 15);
  v16h b;
#pragma unroll
  for (int h = 0; h < 16; ++h) b[h] = (_Float16)X[(size_t)(k0 + kb + h) * n + j];
  return b;
}

// B from LDS (row-major, row stride `ld` halves): B[k][col] = P[(row0+col)*ld + k0 + k]
__device__ __forceinline__ v16h load_b_lds(const _Float16* P, int ld, int row0, int k0) {
  const int lane = laneid();
  const _Float16* p = P + (row0 + (lane & 15)) * ld + k0 + ((lane >> 4) * 16);
  v16h b;
#pragma unroll
  for (int h = 0; h < 16; ++h) b[h] = p[h];
  return b;
}

// ======================= Kernel 1: Q and K projections =======================
__global__ void proj_qk_kernel(const float* __restrict__ xq, const float* __restrict__ xk,
                               const float* __restrict__ Wq, const float* __restrict__ bq,
                               const float* __restrict__ Wk, const float* __restrict__ bk,
                               _Float16* __restrict__ Qh, _Float16* __restrict__ Kh) {
  const int wid = blockIdx.x * (blockDim.x >> 5) + (threadIdx.x >> 5);
  const int half_tiles = Bn * (Nn / 16) * (Dq / 16); // 16384
  const bool isK = wid >= half_tiles;
  const int t = isK ? (wid - half_tiles) : wid;
  const int dt = t & 3;
  const int jt = (t >> 2) & 255;
  const int b  = t >> 10;
  const float* X = (isK ? xk : xq) + (size_t)b * Cc * Nn;
  const float* W = isK ? Wk : Wq;
  const float* bias = isK ? bk : bq;
  _Float16* Out = (isK ? Kh : Qh) + (size_t)b * Nn * Dq;
  const int j0 = jt * 16, d0 = dt * 16;
  const int lane = laneid(), hi = lane >> 4, col = lane & 15;
  const float bb = bias[d0 + col];
  v8f acc;
#pragma unroll
  for (int r = 0; r < 8; ++r) acc[r] = bb;
#pragma unroll
  for (int s = 0; s < Cc / 32; ++s) {
    v16h a = load_a_xT(X, Nn, j0, s * 32);
    v16h w = load_b_wT(W, Cc, d0, s * 32);
    acc = WMMA_F16(a, w, acc);
  }
#pragma unroll
  for (int r = 0; r < 8; ++r)
    Out[(size_t)(j0 + r + 8 * hi) * Dq + d0 + col] = (_Float16)acc[r];
}

// ================= Kernel 2: softmax statistics m_j, 1/Z_j ==================
__global__ void softmax_stats_kernel(const _Float16* __restrict__ Qh,
                                     const _Float16* __restrict__ Kh,
                                     float* __restrict__ mOut, float* __restrict__ rZOut) {
  const int jt = blockIdx.x, b = blockIdx.y;
  const int lane = laneid(), hi = lane >> 4;
  const _Float16* Qb = Qh + (size_t)b * Nn * Dq;
  const _Float16* Kb = Kh + (size_t)b * Nn * Dq;
  const int j0 = jt * 16;
  const v16h a0 = load_a_h(Qb, Dq, j0, 0);
  const v16h a1 = load_a_h(Qb, Dq, j0, 32);
  float mr[8], lr[8];
#pragma unroll
  for (int r = 0; r < 8; ++r) { mr[r] = -3.0e38f; lr[r] = 0.0f; }
  for (int i0 = 0; i0 < Nn; i0 += 16) {
    v16h b0 = load_b_h_kmajor(Kb, Dq, i0, 0);
    v16h b1 = load_b_h_kmajor(Kb, Dq, i0, 32);
    v8f e = {};
    e = WMMA_F16(a0, b0, e);
    e = WMMA_F16(a1, b1, e);
#pragma unroll
    for (int r = 0; r < 8; ++r) {
      const float ev = e[r];
      const float mn = fmaxf(mr[r], ev);
      lr[r] = lr[r] * __expf(mr[r] - mn) + __expf(ev - mn);
      mr[r] = mn;
    }
  }
#pragma unroll
  for (int r = 0; r < 8; ++r) {
#pragma unroll
    for (int off = 1; off < 16; off <<= 1) {
      const float mo = __shfl_xor(mr[r], off, 16);
      const float lo = __shfl_xor(lr[r], off, 16);
      const float mn = fmaxf(mr[r], mo);
      lr[r] = lr[r] * __expf(mr[r] - mn) + lo * __expf(mo - mn);
      mr[r] = mn;
    }
  }
  if ((lane & 15) == 0) {
#pragma unroll
    for (int r = 0; r < 8; ++r) {
      const int j = j0 + r + 8 * hi;
      mOut[(size_t)b * Nn + j]  = mr[r];
      rZOut[(size_t)b * Nn + j] = 1.0f / lr[r];
    }
  }
}

// ============== Kernel 3: V projection with 1/Z_j folded in =================
__global__ void proj_v_kernel(const float* __restrict__ xv, const float* __restrict__ Wv,
                              const float* __restrict__ bv, const float* __restrict__ rZ,
                              _Float16* __restrict__ Vh) {
  const int wid = blockIdx.x * (blockDim.x >> 5) + (threadIdx.x >> 5);
  const int jt = wid & 255;
  const int ct = (wid >> 8) & 15;
  const int b  = wid >> 12;
  const float* X = xv + (size_t)b * Cc * Nn;
  const int j0 = jt * 16, c0 = ct * 16;
  const int lane = laneid(), hi = lane >> 4, col = lane & 15;
  v8f acc;
#pragma unroll
  for (int r = 0; r < 8; ++r) acc[r] = bv[c0 + r + 8 * hi];
#pragma unroll
  for (int s = 0; s < Cc / 32; ++s) {
    v16h a = load_a_w(Wv, Cc, c0, s * 32);
    v16h x = load_b_x(X, Nn, j0, s * 32);
    acc = WMMA_F16(a, x, acc);
  }
  const float rz = rZ[(size_t)b * Nn + j0 + col];
  _Float16* Out = Vh + (size_t)b * Cc * Nn;
#pragma unroll
  for (int r = 0; r < 8; ++r)
    Out[(size_t)(c0 + r + 8 * hi) * Nn + j0 + col] = (_Float16)(acc[r] * rz);
}

// ====================== Kernel 4: fused attend pass =========================
// out[b][c][i] = sum_j Vh[b][c][j] * exp(E[j,i]-m_j).
// Workgroup = (b, 32-wide i tile), 4 waves, each wave owns 64 channels.
// Per 64-j chunk: TDM (wave 0) DMAs V'[256][64] f16 -> LDS while all waves
// compute their 16-j slice of E via WMMA, exponentiate, and stage P in LDS;
// s_wait_tensorcnt + barrier; then V'(16c x 32j) x P(32j x 16i) WMMAs.
__global__ void attend_kernel(const _Float16* __restrict__ Qh, const _Float16* __restrict__ Kh,
                              const _Float16* __restrict__ Vh, const float* __restrict__ m,
                              float* __restrict__ out) {
  __shared__ _Float16 Vt[Cc * JC];          // 32 KB V' chunk
  __shared__ _Float16 Plds[IW * PSTRIDE];   // 4.5 KB P tile, padded rows
  const int i0 = blockIdx.x * IW, b = blockIdx.y;
  const int wave = threadIdx.x >> 5, lane = laneid();
  const int hi = lane >> 4, col = lane & 15;
  const _Float16* Qb = Qh + (size_t)b * Nn * Dq;
  const _Float16* Kb = Kh + (size_t)b * Nn * Dq;
  const _Float16* Vb = Vh + (size_t)b * Cc * Nn;
  const float* mb = m + (size_t)b * Nn;

  // Loop-invariant B operands of the energy WMMAs: K rows i0..i0+31.
  v16h kb[2][2];
#pragma unroll
  for (int it = 0; it < 2; ++it) {
    kb[it][0] = load_b_h_kmajor(Kb, Dq, i0 + it * 16, 0);
    kb[it][1] = load_b_h_kmajor(Kb, Dq, i0 + it * 16, 32);
  }
  const int c0w = wave * 64;
  v8f acc[4][2];
#pragma unroll
  for (int ct = 0; ct < 4; ++ct)
#pragma unroll
    for (int it = 0; it < 2; ++it)
#pragma unroll
      for (int r = 0; r < 8; ++r) acc[ct][it][r] = 0.0f;

#if USE_TDM
  typedef __attribute__((ext_vector_type(4))) unsigned int tdm_v4u;
  typedef __attribute__((ext_vector_type(8))) int tdm_v8i;
  typedef __attribute__((ext_vector_type(4))) int tdm_v4i;
  const unsigned vtAddr = (unsigned)(size_t)(&Vt[0]);   // LDS byte address (addr[31:0])
#endif

  for (int j0 = 0; j0 < Nn; j0 += JC) {
#if USE_TDM
    if (wave == 0) {
      // D# for a [Cc rows][JC cols] f16 tile of Vb starting at column j0.
      const unsigned long long ga = (unsigned long long)(size_t)(Vb + j0);
      const unsigned td0 = (unsigned)(Nn - j0);
      tdm_v4u g0 = { 1u,                                        // count=1, user mode
                     vtAddr,                                    // lds_addr
                     (unsigned)(ga & 0xFFFFFFFFu),              // global_addr lo
                     (unsigned)(((ga >> 32) & 0x01FFFFFFu) | (2u << 30)) }; // hi | type=2
      tdm_v8i g1 = { (int)0x00010000u,                          // data_size=2B
                     (int)((td0 & 0xFFFFu) << 16),              // tensor_dim0 lo16
                     (int)(((td0 >> 16) & 0xFFFFu) | ((unsigned)Cc << 16)), // td0 hi | td1 lo
                     (int)((unsigned)JC << 16),                 // td1 hi=0 | tile_dim0
                     (int)(unsigned)Cc,                         // tile_dim1 | tile_dim2=0
                     (int)(unsigned)Nn,                         // tensor_dim0_stride lo32
                     0, 0 };
      tdm_v4i gz = { 0, 0, 0, 0 };
#if defined(__clang_major__) && (__clang_major__ >= 23)
      tdm_v8i gex = { 0, 0, 0, 0, 0, 0, 0, 0 };
      __builtin_amdgcn_tensor_load_to_lds(g0, g1, gz, gz, gex, 0);
#else
      __builtin_amdgcn_tensor_load_to_lds(g0, g1, gz, gz, 0);
#endif
    }
#else
    // Fallback: cooperative copy of the V' chunk into LDS.
    for (int row = threadIdx.x; row < Cc; row += 128) {
      const uint4* s = (const uint4*)(Vb + (size_t)row * Nn + j0);
      uint4* d = (uint4*)(&Vt[row * JC]);
#pragma unroll
      for (int q = 0; q < JC / 8; ++q) d[q] = s[q];
    }
#endif
    // ---- while the DMA runs: energy tiles + exp -> P in LDS ----
    const int jw = j0 + wave * 16;                // this wave's 16-j slice
    if (j0 + JC < Nn)
      __builtin_prefetch(Qb + (size_t)(jw + JC) * Dq, 0, 1);  // global_prefetch_b8
    {
      v16h a0 = load_a_h(Qb, Dq, jw, 0);
      v16h a1 = load_a_h(Qb, Dq, jw, 32);
      float mvals[8];
#pragma unroll
      for (int r = 0; r < 8; ++r) mvals[r] = mb[jw + r + 8 * hi];
#pragma unroll
      for (int it = 0; it < 2; ++it) {
        v8f e = {};
        e = WMMA_F16(a0, kb[it][0], e);
        e = WMMA_F16(a1, kb[it][1], e);
#pragma unroll
        for (int r = 0; r < 8; ++r) {
          const int jl = wave * 16 + r + 8 * hi;
          const float p = __expf(e[r] - mvals[r]);
          Plds[(it * 16 + col) * PSTRIDE + jl] = (_Float16)p;
        }
      }
    }
#if USE_TDM
    if (wave == 0) __builtin_amdgcn_s_wait_tensorcnt(0);
#endif
    __syncthreads();   // P complete + V' chunk resident in LDS
    // ---- accumulate: out_tile += V'(16c x 32j) x P(32j x 16i) ----
#pragma unroll
    for (int s = 0; s < JC / 32; ++s) {
      v16h pb[2];
#pragma unroll
      for (int it = 0; it < 2; ++it)
        pb[it] = load_b_lds(Plds, PSTRIDE, it * 16, s * 32);
#pragma unroll
      for (int ct = 0; ct < 4; ++ct) {
        v16h a = load_a_h(Vt, JC, c0w + ct * 16, s * 32);
#pragma unroll
        for (int it = 0; it < 2; ++it)
          acc[ct][it] = WMMA_F16(a, pb[it], acc[ct][it]);
      }
    }
    __syncthreads();   // before next chunk overwrites Vt / Plds
  }
  float* ob = out + (size_t)b * Cc * Nn;
#pragma unroll
  for (int ct = 0; ct < 4; ++ct)
#pragma unroll
    for (int it = 0; it < 2; ++it)
#pragma unroll
      for (int r = 0; r < 8; ++r)
        ob[(size_t)(c0w + ct * 16 + r + 8 * hi) * Nn + i0 + it * 16 + col] = acc[ct][it][r];
}

// ============================== launcher ====================================
extern "C" void kernel_launch(void* const* d_in, const int* in_sizes, int n_in,
                              void* d_out, int out_size, void* d_ws, size_t ws_size,
                              hipStream_t stream) {
  (void)in_sizes; (void)n_in; (void)out_size; (void)ws_size;
  const float* xq = (const float*)d_in[0];
  const float* xk = (const float*)d_in[1];
  const float* xv = (const float*)d_in[2];
  const float* Wq = (const float*)d_in[3];
  const float* bq = (const float*)d_in[4];
  const float* Wk = (const float*)d_in[5];
  const float* bk = (const float*)d_in[6];
  const float* Wv = (const float*)d_in[7];
  const float* bv = (const float*)d_in[8];
  float* out = (float*)d_out;

  // Workspace layout (~48.5 MB total):
  //   Qh: 8 MB f16 [B][n][64] | Kh: 8 MB | Vh: 32 MB f16 [B][C][n]
  //   m:  256 KB f32 [B][n]   | rZ: 256 KB f32 [B][n]
  char* ws = (char*)d_ws;
  _Float16* Qh = (_Float16*)(ws);
  _Float16* Kh = (_Float16*)(ws + (size_t)(8u << 20));
  _Float16* Vh = (_Float16*)(ws + (size_t)(16u << 20));
  float* mBuf  = (float*)(ws + (size_t)(48u << 20));
  float* rZ    = (float*)(ws + (size_t)(48u << 20) + (size_t)Bn * Nn * sizeof(float));

  proj_qk_kernel<<<dim3(8192), dim3(128), 0, stream>>>(xq, xk, Wq, bq, Wk, bk, Qh, Kh);
  softmax_stats_kernel<<<dim3(Nn / 16, Bn), dim3(32), 0, stream>>>(Qh, Kh, mBuf, rZ);
  proj_v_kernel<<<dim3(16384), dim3(128), 0, stream>>>(xv, Wv, bv, rZ, Vh);
  attend_kernel<<<dim3(Nn / IW, Bn), dim3(128), 0, stream>>>(Qh, Kh, Vh, mBuf, out);
}
